// GMM_48661979464133
// MI455X (gfx1250) — matrix operand order repaired and checked by reading, not compiled, not run on hardware
//
#include <hip/hip_runtime.h>

// GMM energy: out[n] = -logsumexp_m( log_const[m] - 0.5*(x_n-mu_m)^T S_m (x_n-mu_m) )
// N=65536, K=128, M=8.
// Expanded form: 0.5*(x-mu)^T S (x-mu) = 0.5*x^T S x - b.x + 0.5*c,  b = S mu, c = mu^T S mu.
//   - Precompute kernel: b[m][k], lcp[m] = log_const[m] - 0.5*c_m, and bf16 B-fragment-swizzled S.
//   - Main kernel: A-fragments = bf16(x), built once, register-resident across all components.
//     Q = x @ S_m via v_wmma_f32_16x16x32_bf16; quad = rowsum(x * (0.5*Q - b)) on the
//     accumulator layout with vectorized ds_load_b128 xC reads + shfl_xor reduction.

#define GN 65536
#define GK 128
#define GM 8
#define ROWS_PER_WG 128   // 4 waves x 2 row-tiles x 16 rows
#define THREADS 128

// Workspace layout (bytes)
#define WS_B_OFF    0                         // float b[8][128]
#define WS_LCP_OFF  4096                      // float lcp[8]
#define WS_S_OFF    4608                      // ushort Sswz[8][128*128]
#define WS_NEEDED   (WS_S_OFF + GM * GK * GK * 2)
 
typedef __attribute__((ext_vector_type(16))) __bf16 v16bf;
typedef __attribute__((ext_vector_type(8)))  float  v8f;

__device__ __forceinline__ unsigned short f32_to_bf16_bits(float f) {
  unsigned u = __builtin_bit_cast(unsigned, f);
  unsigned r = u + 0x7FFFu + ((u >> 16) & 1u);   // round-to-nearest-even
  return (unsigned short)(r >> 16);
}
__device__ __forceinline__ __bf16 f32_to_bf16(float f) {
  return __builtin_bit_cast(__bf16, f32_to_bf16_bits(f));
}
__device__ __forceinline__ float bf_lo(unsigned u) {
  return __builtin_bit_cast(float, u << 16);
}
__device__ __forceinline__ float bf_hi(unsigned u) {
  return __builtin_bit_cast(float, u & 0xFFFF0000u);
}

// B-fragment swizzle: frag (t,c), lane l, element e holds S[32c + 16*(l>=16) + e][16t + (l&15)]
__device__ __forceinline__ int s_swizzle_dst(int i /* k*128+j */) {
  const int k = i >> 7, j = i & 127;
  const int c = k >> 5, rem = k & 31, hh = rem >> 4, e = rem & 15;
  const int t = j >> 4;
  return (((t * 4 + c) * 32 + (hh * 16 + (j & 15))) << 4) | e;
}

__global__ __launch_bounds__(128) void gmm_precompute(
    const float* __restrict__ mus, const float* __restrict__ invs,
    const float* __restrict__ lc, float* __restrict__ ws, int doS)
{
  const int m = blockIdx.x;        // GM blocks
  const int k = threadIdx.x;       // GK threads
  const float* Sm  = invs + (size_t)m * GK * GK;
  const float* mum = mus + m * GK;

  float s = 0.f;
  for (int j = 0; j < GK; ++j) s += Sm[k * GK + j] * mum[j];
  ws[m * GK + k] = s;                       // b[m][k]

  __shared__ float red[GK];
  red[k] = s * mum[k];
  __syncthreads();
  for (int off = 64; off > 0; off >>= 1) {
    if (k < off) red[k] += red[k + off];
    __syncthreads();
  }
  if (k == 0) ws[WS_LCP_OFF / 4 + m] = lc[m] - 0.5f * red[0];

  if (doS) {
    unsigned short* sb = (unsigned short*)((char*)ws + WS_S_OFF) + (size_t)m * GK * GK;
    for (int i = k; i < GK * GK; i += 128)
      sb[s_swizzle_dst(i)] = f32_to_bf16_bits(Sm[i]);
  }
}

__global__ __launch_bounds__(THREADS) void gmm_energy_wmma(
    const float* __restrict__ x, const float* __restrict__ invs,
    const float* __restrict__ ws, const unsigned short* __restrict__ Sswz,
    float* __restrict__ out)
{
  const float* bvec = ws;                    // b[8][128]
  const float* lcp  = ws + WS_LCP_OFF / 4;   // lcp[8]

  // 32 KB x in C-accumulator order (bf16) + 32 KB S_m in B-fragment order (bf16)
  __shared__ alignas(16) unsigned short sXC[ROWS_PER_WG * GK];
  __shared__ alignas(32) unsigned short sS[GK * GK];

  const int tid   = threadIdx.x;
  const int lane  = tid & 31;
  const int wave  = tid >> 5;
  const int h     = lane >> 4;
  const int l15   = lane & 15;
  const int row0g = blockIdx.x * ROWS_PER_WG;

  // Stage x into C-layout order: entry ((((w*2+rt)*8+t)*32 + lane)*8 + r)
  // holds x[w*32 + rt*16 + 8*(lane>=16) + r][16t + (lane&15)]  (coalesced global reads)
  for (int i = tid; i < ROWS_PER_WG * GK; i += THREADS) {
    const int row = i >> 7, col = i & 127;
    const int w = row >> 5, rem = row & 31, rt = (rem >> 4) & 1;
    const int h3 = (rem >> 3) & 1, r = rem & 7;
    const int t = col >> 4;
    const int dst = ((((w * 2 + rt) * 8 + t) * 32 + (h3 * 16 + (col & 15))) << 3) | r;
    sXC[dst] = f32_to_bf16_bits(x[(size_t)(row0g + row) * GK + col]);
  }

  // A fragments = bf16(x), built once, live across the whole component loop.
  // A layout (16x32 bf16): lane = sample row (l%16); element e -> K = e + 16h + 32c.
  v16bf A0[4], A1[4];
  {
    const size_t gr0 = (size_t)(row0g + wave * 32 + l15) * GK;
    #pragma unroll
    for (int c = 0; c < 4; ++c) {
      #pragma unroll
      for (int e = 0; e < 16; ++e) {
        const int kk = e + 16 * h + 32 * c;
        A0[c][e] = f32_to_bf16(x[gr0 + kk]);
        A1[c][e] = f32_to_bf16(x[gr0 + 16 * GK + kk]);
      }
    }
  }

  float vmax0[8], vsum0[8], vmax1[8], vsum1[8];
  #pragma unroll
  for (int r = 0; r < 8; ++r) {
    vmax0[r] = -INFINITY; vsum0[r] = 0.f;
    vmax1[r] = -INFINITY; vsum1[r] = 0.f;
  }

  for (int m = 0; m < GM; ++m) {
    __syncthreads();
    if (Sswz) {
      // Pure 16B copy of pre-swizzled bf16 S_m (L2-resident workspace)
      const unsigned short* gs = Sswz + (size_t)m * GK * GK;
      for (int i = tid; i < (GK * GK) / 8; i += THREADS) {
        unsigned tmp[4];
        __builtin_memcpy(tmp, gs + i * 8, 16);
        __builtin_memcpy(&sS[i * 8], tmp, 16);
      }
    } else {
      // Fallback: convert+swizzle from f32 inv_sigmas
      const float* Sm = invs + (size_t)m * GK * GK;
      for (int i = tid; i < GK * GK; i += THREADS)
        sS[s_swizzle_dst(i)] = f32_to_bf16_bits(Sm[i]);
    }
    __syncthreads();

    // Prefetch next component's S (global_prefetch_b8)
    if (m + 1 < GM) {
      if (Sswz) {
        const char* nxt = (const char*)(Sswz + (size_t)(m + 1) * GK * GK);
        __builtin_prefetch(nxt + tid * 128, 0, 1);
        __builtin_prefetch(nxt + tid * 128 + 16384, 0, 1);
      } else {
        const char* nxt = (const char*)(invs + (size_t)(m + 1) * GK * GK);
        #pragma unroll
        for (int p = 0; p < 4; ++p)
          __builtin_prefetch(nxt + tid * 128 + p * 16384, 0, 1);
      }
    }

    float vd0[8], vd1[8];
    #pragma unroll
    for (int r = 0; r < 8; ++r) { vd0[r] = 0.f; vd1[r] = 0.f; }

    for (int t = 0; t < 8; ++t) {
      // Load all 4 B fragments + both xC vectors up front -> single DS wait, 8 WMMAs stream
      v16bf B[4];
      #pragma unroll
      for (int c = 0; c < 4; ++c)
        __builtin_memcpy(&B[c], &sS[(((t * 4 + c) * 32 + lane) << 4)], 32);

      unsigned xc0[4], xc1[4];
      __builtin_memcpy(xc0, &sXC[(((wave * 2 + 0) * 8 + t) * 32 + lane) << 3], 16);
      __builtin_memcpy(xc1, &sXC[(((wave * 2 + 1) * 8 + t) * 32 + lane) << 3], 16);

      const int col = t * 16 + l15;
      const float bv = bvec[m * GK + col];

      v8f acc0 = {}; v8f acc1 = {};
      #pragma unroll
      for (int c = 0; c < 4; ++c) {
        acc0 = __builtin_amdgcn_wmma_f32_16x16x32_bf16(false, A0[c], false, B[c],
                                                       (short)0, acc0, false, false);
        acc1 = __builtin_amdgcn_wmma_f32_16x16x32_bf16(false, A1[c], false, B[c],
                                                       (short)0, acc1, false, false);
      }

      // per-column contribution: x * (0.5*Q - b)
      #pragma unroll
      for (int r = 0; r < 8; ++r) {
        const float xf0 = (r & 1) ? bf_hi(xc0[r >> 1]) : bf_lo(xc0[r >> 1]);
        const float xf1 = (r & 1) ? bf_hi(xc1[r >> 1]) : bf_lo(xc1[r >> 1]);
        vd0[r] += xf0 * (0.5f * acc0[r] - bv);
        vd1[r] += xf1 * (0.5f * acc1[r] - bv);
      }
    }

    // Sum across the 16 lanes of each half-wave
    #pragma unroll
    for (int r = 0; r < 8; ++r) {
      #pragma unroll
      for (int s = 1; s < 16; s <<= 1) {
        vd0[r] += __shfl_xor(vd0[r], s, 32);
        vd1[r] += __shfl_xor(vd1[r], s, 32);
      }
    }

    // Online logsumexp update: V = lcp[m] - (0.5*xSx - b.x)
    const float lcm = lcp[m];
    #pragma unroll
    for (int r = 0; r < 8; ++r) {
      const float v0  = lcm - vd0[r];
      const float nm0 = fmaxf(vmax0[r], v0);
      vsum0[r] = vsum0[r] * __expf(vmax0[r] - nm0) + __expf(v0 - nm0);
      vmax0[r] = nm0;
      const float v1  = lcm - vd1[r];
      const float nm1 = fmaxf(vmax1[r], v1);
      vsum1[r] = vsum1[r] * __expf(vmax1[r] - nm1) + __expf(v1 - nm1);
      vmax1[r] = nm1;
    }
  }

  if (l15 == 0) {
    #pragma unroll
    for (int r = 0; r < 8; ++r) {
      const int rr = row0g + wave * 32 + r + 8 * h;
      out[rr]      = -(vmax0[r] + __logf(vsum0[r]));
      out[rr + 16] = -(vmax1[r] + __logf(vsum1[r]));
    }
  }
}

extern "C" void kernel_launch(void* const* d_in, const int* in_sizes, int n_in,
                              void* d_out, int out_size, void* d_ws, size_t ws_size,
                              hipStream_t stream) {
  (void)in_sizes; (void)n_in; (void)out_size;
  const float* x    = (const float*)d_in[0];
  const float* mus  = (const float*)d_in[1];
  const float* invs = (const float*)d_in[2];
  const float* lc   = (const float*)d_in[3];
  float* out = (float*)d_out;
  float* ws  = (float*)d_ws;

  const int doS = (ws_size >= (size_t)WS_NEEDED) ? 1 : 0;
  gmm_precompute<<<GM, 128, 0, stream>>>(mus, invs, lc, ws, doS);

  const unsigned short* Sswz =
      doS ? (const unsigned short*)((const char*)d_ws + WS_S_OFF) : nullptr;
  gmm_energy_wmma<<<GN / ROWS_PER_WG, THREADS, 0, stream>>>(x, invs, ws, Sswz, out);
}
